// SMDAttentionHead_29308856827973
// MI455X (gfx1250) — compile-verified, matrix operand order checked by reading, not run on hardware
//
#include <hip/hip_runtime.h>
#include <hip/hip_bf16.h>

typedef __attribute__((ext_vector_type(16))) __bf16 v16bf;
typedef __attribute__((ext_vector_type(8)))  __bf16 v8bf;
typedef __attribute__((ext_vector_type(8)))  float  v8f;

#define BATCH 4
#define SEQ   2048
#define FDIM  1024
#define EDIM  128
#define BN    (BATCH * SEQ)   // 8192 rows total

__device__ inline v16bf cat8(v8bf lo, v8bf hi) {
    return __builtin_shufflevector(lo, hi, 0,1,2,3,4,5,6,7,8,9,10,11,12,13,14,15);
}

// ---- block reductions (256 threads = 8 wave32) -----------------------------
__device__ inline float blockMax(float v, float* red) {
#pragma unroll
    for (int off = 16; off > 0; off >>= 1) v = fmaxf(v, __shfl_xor(v, off, 32));
    int w = threadIdx.x >> 5;
    if ((threadIdx.x & 31) == 0) red[w] = v;
    __syncthreads();
    float r = red[0];
#pragma unroll
    for (int i = 1; i < 8; i++) r = fmaxf(r, red[i]);
    __syncthreads();
    return r;
}

__device__ inline float blockSum(float v, float* red) {
#pragma unroll
    for (int off = 16; off > 0; off >>= 1) v += __shfl_xor(v, off, 32);
    int w = threadIdx.x >> 5;
    if ((threadIdx.x & 31) == 0) red[w] = v;
    __syncthreads();
    float r = red[0];
#pragma unroll
    for (int i = 1; i < 8; i++) r += red[i];
    __syncthreads();
    return r;
}

// ---- W [F,E] f32 -> Wt [E,F] bf16 ------------------------------------------
__global__ void wt_convert(const float* __restrict__ W, __bf16* __restrict__ Wt) {
    int id = blockIdx.x * 256 + threadIdx.x;      // EDIM*FDIM = 131072
    int e = id >> 10, k = id & 1023;
    Wt[id] = (__bf16)W[k * EDIM + e];
}

// ---- projections: Q/K = feat @ W, bf16 WMMA 16x16x32, one wave per 16x128 tile
__global__ void proj_kernel(const float* __restrict__ feat,
                            const __bf16* __restrict__ Wqt,
                            const __bf16* __restrict__ Wkt,
                            __bf16* __restrict__ Q, __bf16* __restrict__ K) {
    int wave = (blockIdx.x * blockDim.x + threadIdx.x) >> 5;  // 1024 waves
    int lane = threadIdx.x & 31;
    int sel  = wave & 1;
    int tile = wave >> 1;                                     // 0..511
    const __bf16* Wt  = sel ? Wkt : Wqt;
    __bf16*       Out = sel ? K   : Q;
    int row0 = tile * 16;
    int m = lane & 15, half = lane >> 4;
    const float* arow = feat + (size_t)(row0 + m) * FDIM;

    v8f acc[8] = {};   // 8 e-tiles of 16 columns each
    for (int k0 = 0; k0 < FDIM; k0 += 32) {
        int ka = k0 + half * 8;
        // A tile: convert f32 features -> bf16 per ISA A layout
        float4 f0 = *(const float4*)(arow + ka);
        float4 f1 = *(const float4*)(arow + ka + 4);
        float4 f2 = *(const float4*)(arow + ka + 16);
        float4 f3 = *(const float4*)(arow + ka + 20);
        v16bf A;
        A[0]=(__bf16)f0.x; A[1]=(__bf16)f0.y; A[2]=(__bf16)f0.z; A[3]=(__bf16)f0.w;
        A[4]=(__bf16)f1.x; A[5]=(__bf16)f1.y; A[6]=(__bf16)f1.z; A[7]=(__bf16)f1.w;
        A[8]=(__bf16)f2.x; A[9]=(__bf16)f2.y; A[10]=(__bf16)f2.z; A[11]=(__bf16)f2.w;
        A[12]=(__bf16)f3.x; A[13]=(__bf16)f3.y; A[14]=(__bf16)f3.z; A[15]=(__bf16)f3.w;
#pragma unroll
        for (int e = 0; e < 8; e++) {
            const __bf16* wrow = Wt + (size_t)(e * 16 + m) * FDIM + ka;
            v16bf Bm = cat8(*(const v8bf*)wrow, *(const v8bf*)(wrow + 16));
            acc[e] = __builtin_amdgcn_wmma_f32_16x16x32_bf16(
                false, A, false, Bm, (short)0, acc[e], false, false);
        }
    }
#pragma unroll
    for (int e = 0; e < 8; e++)
#pragma unroll
        for (int r = 0; r < 8; r++)
            Out[(size_t)(row0 + r + half * 8) * EDIM + e * 16 + m] = (__bf16)acc[e][r];
}

// ---- charge0 = sigmoid(feat . charge_w + b); one wave per row --------------
__global__ void charge_init(const float* __restrict__ feat,
                            const float* __restrict__ cw,
                            const float* __restrict__ cb,
                            float* __restrict__ c0, float* __restrict__ received) {
    int wave = (blockIdx.x * 256 + threadIdx.x) >> 5;   // 8192 waves
    int lane = threadIdx.x & 31;
    const float* fr = feat + (size_t)wave * FDIM;
    float acc = 0.f;
    for (int k = lane; k < FDIM; k += 32) acc = fmaf(fr[k], cw[k], acc);
#pragma unroll
    for (int off = 16; off > 0; off >>= 1) acc += __shfl_xor(acc, off, 32);
    if (lane == 0) {
        c0[wave] = 1.f / (1.f + __expf(-(acc + cb[0])));
        received[wave] = 0.f;
    }
}

// ---- base = Q.K^T/sqrt(E) + locality -> d_out; one wave per 16x16 tile -----
__global__ void base_kernel(const __bf16* __restrict__ Q, const __bf16* __restrict__ K,
                            float* __restrict__ base, const float* __restrict__ ls_p) {
    int wave = (blockIdx.x * blockDim.x + threadIdx.x) >> 5;  // 65536 waves
    int lane = threadIdx.x & 31;
    int b  = wave >> 14;
    int ti = (wave >> 7) & 127, tj = wave & 127;
    int i0 = ti * 16, j0 = tj * 16;
    const __bf16* Qb = Q + (size_t)b * SEQ * EDIM;
    const __bf16* Kb = K + (size_t)b * SEQ * EDIM;
    int m = lane & 15, half = lane >> 4;

    v8f acc = {};
#pragma unroll
    for (int k0 = 0; k0 < EDIM; k0 += 32) {
        int ka = k0 + half * 8;
        const __bf16* qa = Qb + (size_t)(i0 + m) * EDIM + ka;
        const __bf16* kb = Kb + (size_t)(j0 + m) * EDIM + ka;
        v16bf A  = cat8(*(const v8bf*)qa, *(const v8bf*)(qa + 16));
        v16bf Bm = cat8(*(const v8bf*)kb, *(const v8bf*)(kb + 16));
        acc = __builtin_amdgcn_wmma_f32_16x16x32_bf16(
            false, A, false, Bm, (short)0, acc, false, false);
    }
    float ls = ls_p[0];
    const float scale = 0.08838834764831845f;  // 1/sqrt(128)
    int j = j0 + m;
#pragma unroll
    for (int r = 0; r < 8; r++) {
        int i = i0 + r + half * 8;
        float d = fmaxf(fabsf((float)(i - j)), 1.0f);
        base[((size_t)b * SEQ + i) * SEQ + j] = acc[r] * scale + ls / d;
    }
}

// ---- one softmax pass: received[b,j] += sum_i softmax(base*m)_{i,j} --------
// multiplier m_{ij} = 1 + step * sum_{c<NC} c_s[i]*c_s[j]; base stays in L2.
template <int NC>
__global__ void step_kernel(const float* __restrict__ base,
                            const float* __restrict__ charges,  // [NC][BN]
                            float* __restrict__ received,
                            const float* __restrict__ step_p) {
    __shared__ float colsum[SEQ];
    __shared__ float red[8];
    int t = threadIdx.x;
    int row0 = blockIdx.x * 16;       // 16 rows per block
    int b = row0 >> 11;
    float stepv = step_p[0];
#pragma unroll
    for (int q = 0; q < 8; q++) colsum[t + 256 * q] = 0.f;
    float cc[NC > 0 ? NC : 1][8];
#pragma unroll
    for (int c = 0; c < NC; c++)
#pragma unroll
        for (int q = 0; q < 8; q++)
            cc[c][q] = charges[c * BN + b * SEQ + t + 256 * q];
    __syncthreads();

    for (int r = 0; r < 16; r++) {
        int row = row0 + r;
        const float* brow = base + (size_t)row * SEQ;
        float rc[NC > 0 ? NC : 1];
#pragma unroll
        for (int c = 0; c < NC; c++) rc[c] = charges[c * BN + row];
        float l[8], lmax = -__builtin_inff();
#pragma unroll
        for (int q = 0; q < 8; q++) {
            float mult = 1.f;
#pragma unroll
            for (int c = 0; c < NC; c++) mult = fmaf(stepv * rc[c], cc[c][q], mult);
            float v = brow[t + 256 * q] * mult;
            l[q] = v; lmax = fmaxf(lmax, v);
        }
        lmax = blockMax(lmax, red);
        float s = 0.f;
#pragma unroll
        for (int q = 0; q < 8; q++) { l[q] = __expf(l[q] - lmax); s += l[q]; }
        s = blockSum(s, red);
        float inv = 1.f / s;
#pragma unroll
        for (int q = 0; q < 8; q++) colsum[t + 256 * q] += l[q] * inv;
    }
    __syncthreads();
#pragma unroll
    for (int q = 0; q < 8; q++)
        atomicAdd(&received[b * SEQ + t + 256 * q], colsum[t + 256 * q]);
}

// ---- charge_{s} = charge_{s-1} * (1 - decay*sigmoid(received-1)); reset recv
__global__ void charge_update(const float* __restrict__ cprev, float* __restrict__ cnew,
                              float* __restrict__ received, const float* __restrict__ dec_p) {
    int id = blockIdx.x * 256 + threadIdx.x;   // BN
    float rr = received[id];
    float sig = 1.f / (1.f + __expf(-(rr - 1.f)));
    cnew[id] = cprev[id] * (1.f - dec_p[0] * sig);
    received[id] = 0.f;
}

// ---- final: out = softmax(base * m4), in place over base (one row/block) ---
__global__ void final_kernel(float* __restrict__ base,
                             const float* __restrict__ charges,  // [4][BN]
                             const float* __restrict__ step_p) {
    __shared__ float red[8];
    int t = threadIdx.x;
    int row = blockIdx.x;
    int b = row >> 11;
    float stepv = step_p[0];
    float* brow = base + (size_t)row * SEQ;
    float rc[4];
#pragma unroll
    for (int c = 0; c < 4; c++) rc[c] = charges[c * BN + row];
    float l[8], lmax = -__builtin_inff();
#pragma unroll
    for (int q = 0; q < 8; q++) {
        int j = t + 256 * q;
        float mult = 1.f;
#pragma unroll
        for (int c = 0; c < 4; c++) mult = fmaf(stepv * rc[c], charges[c * BN + b * SEQ + j], mult);
        float v = brow[j] * mult;
        l[q] = v; lmax = fmaxf(lmax, v);
    }
    lmax = blockMax(lmax, red);
    float s = 0.f;
#pragma unroll
    for (int q = 0; q < 8; q++) { l[q] = __expf(l[q] - lmax); s += l[q]; }
    s = blockSum(s, red);
    float inv = 1.f / s;
#pragma unroll
    for (int q = 0; q < 8; q++) brow[t + 256 * q] = l[q] * inv;
}

extern "C" void kernel_launch(void* const* d_in, const int* in_sizes, int n_in,
                              void* d_out, int out_size, void* d_ws, size_t ws_size,
                              hipStream_t stream) {
    const float* feat = (const float*)d_in[0];
    const float* Wq   = (const float*)d_in[1];
    const float* Wk   = (const float*)d_in[2];
    const float* cw   = (const float*)d_in[3];
    const float* cb   = (const float*)d_in[4];
    const float* ls   = (const float*)d_in[5];
    const float* stp  = (const float*)d_in[6];
    const float* dec  = (const float*)d_in[7];
    float* out = (float*)d_out;
    char*  ws  = (char*)d_ws;

    __bf16* Wqt = (__bf16*)(ws);                         // 256 KB
    __bf16* Wkt = (__bf16*)(ws + 262144);                // 256 KB
    __bf16* Qb  = (__bf16*)(ws + 524288);                // 2 MB
    __bf16* Kb  = (__bf16*)(ws + 2621440);               // 2 MB
    float*  ch  = (float*)(ws + 4718592);                // [5][BN] charges
    float*  received = (float*)(ws + 4718592 + 5 * BN * 4);
    float* c0 = ch;
    float* c1 = ch + BN;

    wt_convert<<<512, 256, 0, stream>>>(Wq, Wqt);
    wt_convert<<<512, 256, 0, stream>>>(Wk, Wkt);
    proj_kernel<<<128, 256, 0, stream>>>(feat, Wqt, Wkt, Qb, Kb);
    charge_init<<<1024, 256, 0, stream>>>(feat, cw, cb, c0, received);
    base_kernel<<<8192, 256, 0, stream>>>(Qb, Kb, out, ls);

    step_kernel<0><<<512, 256, 0, stream>>>(out, c1, received, stp);
    charge_update<<<32, 256, 0, stream>>>(c0, c1, received, dec);
    step_kernel<1><<<512, 256, 0, stream>>>(out, c1, received, stp);
    charge_update<<<32, 256, 0, stream>>>(c1, ch + 2 * BN, received, dec);
    step_kernel<2><<<512, 256, 0, stream>>>(out, c1, received, stp);
    charge_update<<<32, 256, 0, stream>>>(ch + 2 * BN, ch + 3 * BN, received, dec);
    step_kernel<3><<<512, 256, 0, stream>>>(out, c1, received, stp);
    charge_update<<<32, 256, 0, stream>>>(ch + 3 * BN, ch + 4 * BN, received, dec);

    final_kernel<<<8192, 256, 0, stream>>>(out, c1, stp);
}
